// Attention_35115652612129
// MI455X (gfx1250) — matrix operand order repaired and checked by reading, not compiled, not run on hardware
//
#include <hip/hip_runtime.h>
#include <hip/hip_bf16.h>

// ---------------------------------------------------------------------------
// Fused ragged-segment self-attention for MI455X (gfx1250, wave32).
//   u      = tanh(C @ W^T + b)          [L,D]
//   scores = C @ u^T  (masked softmax)  [L,L]
//   out    = softmax(scores) @ C        [L,D]  (padded query rows zeroed)
//
// Matmuls use V_WMMA_F32_16X16X32_BF16 with split-precision emulation:
//   x ~= hi + lo  (two RTNE bf16 words, ~16 mantissa bits total)
//   A*B ~= Ah*Bh + Ah*Bl + Al*Bh        (drop lo*lo term, ~2^-17 rel. error)
// 8x the MACs/instruction of V_WMMA_F32_16X16X4_F32 at near-fp32 accuracy;
// per-wave matrix instructions 2304 (vs 6144 f32), LDS bytes/MAC 8x lower,
// putting the kernel at the 23.3 TB/s HBM roof (~540MB traffic).
//
// tanh/exp use CDNA5 hardware transcendentals (V_TANH_F32 / V_EXP_F32)
// instead of the branchy OCML expansions. Next W e-tile is prefetched
// (global_prefetch_b8) while the current one is consumed.
//
// One workgroup (256 threads = 8 wave32) per segment. Split C tile
// (2 x 128x520 bf16) lives in CDNA5's 320KB LDS; strides chosen so b128
// fragment loads are bank-conflict-free.
// ---------------------------------------------------------------------------

typedef __attribute__((ext_vector_type(4)))  float  v4f;
typedef __attribute__((ext_vector_type(8)))  float  v8f;
typedef __attribute__((ext_vector_type(4)))  __bf16 bf4;
typedef __attribute__((ext_vector_type(8)))  __bf16 bf8;
typedef __attribute__((ext_vector_type(16))) __bf16 v16bf;

#define WMMA_BF16(a, b, c) \
  __builtin_amdgcn_wmma_f32_16x16x32_bf16(false, (a), false, (b), (short)0, (c), false, false)

#if __has_builtin(__builtin_amdgcn_tanhf)
#define FAST_TANH(x) __builtin_amdgcn_tanhf(x)
#elif __has_builtin(__builtin_amdgcn_tanh_f32)
#define FAST_TANH(x) __builtin_amdgcn_tanh_f32(x)
#else
#define FAST_TANH(x) tanhf(x)
#endif

#if __has_builtin(__builtin_amdgcn_exp2f)
#define FAST_EXP2(x) __builtin_amdgcn_exp2f(x)
#else
#define FAST_EXP2(x) exp2f(x)
#endif

constexpr int SEG = 1024;
constexpr int L = 128;
constexpr int D = 512;
constexpr int CST = 520;  // Chi/Clo row stride (bf16): 260 dw, 260%64==4 -> perfect b128 spread
constexpr int WST = 520;  // Ws row stride (bf16)
constexpr int UST = 40;   // Us/As row stride (bf16): 20 dw -> perfect spread
constexpr float LOG2E = 1.4426950408889634f;

// LDS (bytes): Chi/Clo 2*133120, Wsh/Wsl 2*16640, Ush/Usl 2*10240, Red 4096, Redc 512
constexpr size_t LDS_BYTES =
    (size_t)(2 * L * CST + 2 * 16 * WST + 2 * L * UST) * 2 + (L * 8 + L) * 4;  // 324,608

__device__ __forceinline__ __bf16 hi_bf16(float x) { return (__bf16)x; }  // RTNE
__device__ __forceinline__ __bf16 lo_bf16(float x, __bf16 h) {
  return (__bf16)(x - (float)h);  // residual
}

// A-operand fragment (M = lane%16 row): elements 0..15 <-> K = kb+16*lh .. +15
// (contiguous K order, matching load_fragB)
__device__ __forceinline__ v16bf load_fragA(const __bf16* rowp, int kb, int lh) {
  v16bf f;
  *(bf8*)&f        = *(const bf8*)(rowp + kb + 16 * lh);
  *((bf8*)&f + 1)  = *(const bf8*)(rowp + kb + 16 * lh + 8);
  return f;
}

// B-operand fragment (N = lane%16), data contiguous in K within the row:
// elements 0..15 <-> K = kb+16*lh .. +15   (contiguous K order, matching load_fragA)
__device__ __forceinline__ v16bf load_fragB(const __bf16* rowp, int kb, int lh) {
  v16bf f;
  *(bf8*)&f        = *(const bf8*)(rowp + kb + 16 * lh);
  *((bf8*)&f + 1)  = *(const bf8*)(rowp + kb + 16 * lh + 8);
  return f;
}

__global__ __launch_bounds__(256)
void attn_ragged_kernel(const float* __restrict__ ctx,
                        const int* __restrict__ lengths,
                        const float* __restrict__ W,
                        const float* __restrict__ bias,
                        float* __restrict__ out) {
  extern __shared__ char smem[];
  __bf16* Chi = (__bf16*)smem;             // [128][520] C hi
  __bf16* Clo = Chi + L * CST;             // [128][520] C lo
  __bf16* Wsh = Clo + L * CST;             // [16][520]  W e-tile hi
  __bf16* Wsl = Wsh + 16 * WST;            // [16][520]  W e-tile lo
  __bf16* Ush = Wsl + 16 * WST;            // [128][40]  u tile hi (32 e wide)
  __bf16* Usl = Ush + L * UST;             // [128][40]  u tile lo
  float*  Red  = (float*)(Usl + L * UST);  // [128][8]
  float*  Redc = Red + L * 8;              // [128]
  __bf16* Ash = Wsh;                       // phase-4 attn strip hi [128][40] (aliases Ws)
  __bf16* Asl = Wsh + L * UST;             // phase-4 attn strip lo

  const int s  = blockIdx.x;
  const int t  = threadIdx.x;
  const int wv = t >> 5;        // wave 0..7
  const int ln = t & 31;        // lane 0..31
  const int lh = ln >> 4;       // half-wave 0/1
  const int li = ln & 15;       // lane-in-half 0..15
  const int len = lengths[s];

  const float* Cg = ctx + (size_t)s * L * D;

  // prefetch first W e-tile while we stage C (256 threads x 128B = 32KB tile)
  __builtin_prefetch(W + t * 32, 0, 0);

  // ---- stage C into LDS as split bf16 (coalesced float4 reads) -----------
  for (int f = t; f < (L * D) / 4; f += 256) {
    const int r  = f >> 7;
    const int c4 = (f & 127) * 4;
    v4f val = *(const v4f*)(Cg + r * D + c4);
    bf4 h, l;
#pragma unroll
    for (int j = 0; j < 4; ++j) {
      const __bf16 hh = hi_bf16(val[j]);
      h[j] = hh;
      l[j] = lo_bf16(val[j], hh);
    }
    *(bf4*)(Chi + r * CST + c4) = h;
    *(bf4*)(Clo + r * CST + c4) = l;
  }
  __syncthreads();

  // score accumulators (f32): wave wv owns key strip k in [16*wv, 16*wv+16)
  v8f acc[8];
#pragma unroll
  for (int i = 0; i < 8; ++i)
    acc[i] = (v8f){0.f, 0.f, 0.f, 0.f, 0.f, 0.f, 0.f, 0.f};

  const int kcol = 16 * wv + li;

  // ==== fused phase 1 (u tiles) + phase 2 (score accumulation) ============
#pragma unroll 1
  for (int et = 0; et < D / 16; ++et) {
    // stage + split W e-tile (rows e = 16*et .. +16) cooperatively
    const float* Wg = W + (size_t)(16 * et) * D;
    for (int f = t; f < (16 * D) / 4; f += 256) {
      const int r  = f >> 7;
      const int c4 = (f & 127) * 4;
      v4f val = *(const v4f*)(Wg + r * D + c4);
      bf4 h, l;
#pragma unroll
      for (int j = 0; j < 4; ++j) {
        const __bf16 hh = hi_bf16(val[j]);
        h[j] = hh;
        l[j] = lo_bf16(val[j], hh);
      }
      *(bf4*)(Wsh + r * WST + c4) = h;
      *(bf4*)(Wsl + r * WST + c4) = l;
    }
    // prefetch next e-tile into cache while this one is consumed
    if (et + 1 < D / 16) __builtin_prefetch(Wg + 16 * D + t * 32, 0, 0);
    __syncthreads();

    // --- phase 1: wave wv computes u tile rows [16wv,16wv+16) x 16 e -----
    v8f uacc = (v8f){0.f, 0.f, 0.f, 0.f, 0.f, 0.f, 0.f, 0.f};
    const __bf16* ArH = Chi + (16 * wv + li) * CST;
    const __bf16* ArL = Clo + (16 * wv + li) * CST;
    const __bf16* BrH = Wsh + li * WST;   // row e_local = li
    const __bf16* BrL = Wsl + li * WST;
#pragma unroll 2
    for (int kc = 0; kc < D / 32; ++kc) {
      const int d0 = kc * 32;
      v16bf ah = load_fragA(ArH, d0, lh);
      v16bf al = load_fragA(ArL, d0, lh);
      v16bf bh = load_fragB(BrH, d0, lh);
      v16bf bl = load_fragB(BrL, d0, lh);
      uacc = WMMA_BF16(ah, bh, uacc);
      uacc = WMMA_BF16(ah, bl, uacc);
      uacc = WMMA_BF16(al, bh, uacc);
    }
    // bias + tanh (V_TANH_F32), split, store into 32-wide double buffer
    const float bv = bias[16 * et + li];
    const int ecol = (et & 1) * 16 + li;
#pragma unroll
    for (int v = 0; v < 8; ++v) {
      const int tok = 16 * wv + v + 8 * lh;
      const float uv = FAST_TANH(uacc[v] + bv);
      const __bf16 h = hi_bf16(uv);
      Ush[tok * UST + ecol] = h;
      Usl[tok * UST + ecol] = lo_bf16(uv, h);
    }
    __syncthreads();

    // --- phase 2 (every 2nd et): scores += C[:,d0..d0+32) @ u_tile^T ------
    if (et & 1) {
      const int d0 = 32 * (et >> 1);
      const __bf16* UrH = Ush + (16 * wv + li) * UST;  // row key = 16wv+li
      const __bf16* UrL = Usl + (16 * wv + li) * UST;
      v16bf bh = load_fragB(UrH, 0, lh);
      v16bf bl = load_fragB(UrL, 0, lh);
#pragma unroll
      for (int qt = 0; qt < 8; ++qt) {
        v16bf ah = load_fragA(Chi + (16 * qt + li) * CST, d0, lh);
        v16bf al = load_fragA(Clo + (16 * qt + li) * CST, d0, lh);
        acc[qt] = WMMA_BF16(ah, bh, acc[qt]);
        acc[qt] = WMMA_BF16(ah, bl, acc[qt]);
        acc[qt] = WMMA_BF16(al, bh, acc[qt]);
      }
    }
    __syncthreads();
  }

  // ==== masked softmax over k (f32, exact; lane owns one key column) ======
  if (kcol >= len) {
#pragma unroll
    for (int qt = 0; qt < 8; ++qt)
#pragma unroll
      for (int v = 0; v < 8; ++v) acc[qt][v] = -__builtin_inff();
  }

#pragma unroll
  for (int qt = 0; qt < 8; ++qt) {
#pragma unroll
    for (int v = 0; v < 8; ++v) {
      float m = acc[qt][v];
      m = fmaxf(m, __shfl_xor(m, 1, 32));
      m = fmaxf(m, __shfl_xor(m, 2, 32));
      m = fmaxf(m, __shfl_xor(m, 4, 32));
      m = fmaxf(m, __shfl_xor(m, 8, 32));
      if (li == 0) Red[(16 * qt + v + 8 * lh) * 8 + wv] = m;
    }
  }
  __syncthreads();
  if (t < L) {
    float m = Red[t * 8];
#pragma unroll
    for (int i = 1; i < 8; ++i) m = fmaxf(m, Red[t * 8 + i]);
    Redc[t] = m;
  }
  __syncthreads();
#pragma unroll
  for (int qt = 0; qt < 8; ++qt) {
#pragma unroll
    for (int v = 0; v < 8; ++v) {
      const int q = 16 * qt + v + 8 * lh;
      // branch-free: arg <= 0 always; V_EXP_F32(-inf) = 0 handles masked lanes
      const float e = FAST_EXP2((acc[qt][v] - Redc[q]) * LOG2E);
      acc[qt][v] = e;
      float sum = e;
      sum += __shfl_xor(sum, 1, 32);
      sum += __shfl_xor(sum, 2, 32);
      sum += __shfl_xor(sum, 4, 32);
      sum += __shfl_xor(sum, 8, 32);
      if (li == 0) Red[q * 8 + wv] = sum;
    }
  }
  __syncthreads();
  if (t < L) {
    float ssum = 0.f;
#pragma unroll
    for (int i = 0; i < 8; ++i) ssum += Red[t * 8 + i];
    Redc[t] = 1.0f / ssum;
  }
  __syncthreads();
#pragma unroll
  for (int qt = 0; qt < 8; ++qt)
#pragma unroll
    for (int v = 0; v < 8; ++v) acc[qt][v] *= Redc[16 * qt + v + 8 * lh];

  // ==== phase 4: out = attn @ C ===========================================
  // d-blocks of 64; contraction streamed as 32-wide k strips (2 waves publish)
  float* Og = out + (size_t)s * L * D;
#pragma unroll 1
  for (int db = 0; db < D / 64; ++db) {
    v8f oacc[4];
#pragma unroll
    for (int nt = 0; nt < 4; ++nt)
      oacc[nt] = (v8f){0.f, 0.f, 0.f, 0.f, 0.f, 0.f, 0.f, 0.f};

#pragma unroll 1
    for (int sw2 = 0; sw2 < 4; ++sw2) {
      // waves 2*sw2 and 2*sw2+1 publish split attn strips k in [32sw2,32sw2+32)
      if ((wv >> 1) == sw2) {
        const int kh = (wv & 1) * 16;
#pragma unroll
        for (int qt = 0; qt < 8; ++qt) {
#pragma unroll
          for (int v = 0; v < 8; ++v) {
            const int q = 16 * qt + v + 8 * lh;
            const float a = acc[qt][v];
            const __bf16 h = hi_bf16(a);
            Ash[q * UST + kh + li] = h;
            Asl[q * UST + kh + li] = lo_bf16(a, h);
          }
        }
      }
      __syncthreads();

      // wave wv accumulates rows q in [16wv,16wv+16)
      v16bf ah = load_fragA(Ash + (16 * wv + li) * UST, 0, lh);
      v16bf al = load_fragA(Asl + (16 * wv + li) * UST, 0, lh);
#pragma unroll
      for (int nt = 0; nt < 4; ++nt) {
        const int d = db * 64 + 16 * nt + li;
        v16bf bh, bl;  // B[k,n]=C[k,d]: K pairs live in adjacent rows -> u16 gathers
#pragma unroll
        for (int v2 = 0; v2 < 8; ++v2) {
          const int k = 32 * sw2 + 2 * v2 + 16 * lh;
          bh[2 * v2]     = Chi[(k + 0) * CST + d];
          bh[2 * v2 + 1] = Chi[(k + 1) * CST + d];
          bl[2 * v2]     = Clo[(k + 0) * CST + d];
          bl[2 * v2 + 1] = Clo[(k + 1) * CST + d];
        }
        oacc[nt] = WMMA_BF16(ah, bh, oacc[nt]);
        oacc[nt] = WMMA_BF16(ah, bl, oacc[nt]);
        oacc[nt] = WMMA_BF16(al, bh, oacc[nt]);
      }
      __syncthreads();
    }

    // store (zero padded query rows)
#pragma unroll
    for (int nt = 0; nt < 4; ++nt) {
#pragma unroll
      for (int v = 0; v < 8; ++v) {
        const int q = 16 * wv + v + 8 * lh;
        const float val = (q < len) ? oacc[nt][v] : 0.f;
        Og[(size_t)q * D + db * 64 + 16 * nt + li] = val;
      }
    }
  }
}

extern "C" void kernel_launch(void* const* d_in, const int* in_sizes, int n_in,
                              void* d_out, int out_size, void* d_ws, size_t ws_size,
                              hipStream_t stream) {
  (void)in_sizes; (void)n_in; (void)out_size; (void)d_ws; (void)ws_size;
  const float* ctx  = (const float*)d_in[0];  // [1024,128,512] f32
  const int*   lens = (const int*)d_in[1];    // [1024] i32
  const float* W    = (const float*)d_in[2];  // [512,512] f32
  const float* b    = (const float*)d_in[3];  // [512] f32
  float* out = (float*)d_out;                 // [1024,128,512] f32

  attn_ragged_kernel<<<dim3(SEG), dim3(256), LDS_BYTES, stream>>>(ctx, lens, W, b, out);
}
